// JittedGnnModel_18124761989846
// MI455X (gfx1250) — matrix operand order repaired and checked by reading, not compiled, not run on hardware
//
#include <hip/hip_runtime.h>
#include <hip/hip_bf16.h>

// ---------------------------------------------------------------------------
// GCN (2-layer, PyG-style) for MI455X / gfx1250.
// GEMMs: v_wmma_f32_16x16x32_bf16, B fragments register-resident, 10 M-tiles
// per block, 2-way tile interleave (two accumulators) to kill WMMA->WMMA
// hazard nops, unguarded clause-able stores. Layer-1 bias+ReLU fused into
// layer-2 GEMM A-staging. Message passing fp32, L2-resident.
// ---------------------------------------------------------------------------

#define FEAT 128
#define TSTEPS 12
#define ROWLEN (TSTEPS * FEAT)   // 1536 floats per node
#define LDSW 132                 // padded LDS row stride (shorts): 264B
#define GT 10                    // M tiles per block (GROWS must divide M)
#define GROWS (GT * 16)          // 160 rows per block; 120000 % 160 == 0

typedef __attribute__((ext_vector_type(16))) __bf16 v16bf;
typedef __attribute__((ext_vector_type(8)))  float  v8f;

union FragAB { v16bf v; unsigned u[8]; };
union FragC  { v8f   v; float    f[8]; };

static __device__ __forceinline__ unsigned short f2bf(float f) {
    unsigned u = __builtin_bit_cast(unsigned, f);
    unsigned r = (u + 0x7FFFu + ((u >> 16) & 1u)) >> 16;  // RNE truncate
    return (unsigned short)r;
}

// ---------------------------------------------------------------------------
// utility kernels
// ---------------------------------------------------------------------------
__global__ void zero_f32(float* __restrict__ p, int n) {
    int i = blockIdx.x * blockDim.x + threadIdx.x;
    if (i < n) p[i] = 0.0f;
}

__global__ void degree_kernel(const int* __restrict__ ei, int E, int nNodes,
                              float* __restrict__ deg) {
    int i = blockIdx.x * blockDim.x + threadIdx.x;
    int total = E + nNodes;
    if (i >= total) return;
    int d = (i < E) ? ei[E + i] : (i - E);   // dst row, or self-loop node
    atomicAdd(&deg[d], 1.0f);
}

__global__ void rsqrt_kernel(float* __restrict__ deg, int n) {
    int i = blockIdx.x * blockDim.x + threadIdx.x;
    if (i >= n) return;
    float d = deg[i];
    deg[i] = (d > 0.0f) ? rsqrtf(d) : 0.0f;
}

// W[k][n] fp32 -> Wt[n][k] bf16 (col-major so K-pairs are contiguous)
__global__ void convert_w(const float* __restrict__ W,
                          unsigned short* __restrict__ Wt) {
    int i = blockIdx.x * blockDim.x + threadIdx.x;
    if (i >= FEAT * FEAT) return;
    int k = i >> 7, n = i & 127;
    Wt[n * FEAT + k] = f2bf(W[i]);
}

__global__ void bias_relu(float* __restrict__ y, const float* __restrict__ b,
                          int total) {
    int i = blockIdx.x * blockDim.x + threadIdx.x;
    if (i >= total) return;
    float v = y[i] + b[i & 127];
    y[i] = v > 0.0f ? v : 0.0f;
}

// ---------------------------------------------------------------------------
// GEMM: C[M x 128] = act(A)[M x 128] * W[128 x 128] via WMMA bf16.
// Block: 256 threads (8 waves), 160 rows (10 x 16-row M tiles).
// Wave w owns N-tile w; its 4 B fragments live in registers across all tiles.
// Tiles processed in interleaved pairs -> independent alternating WMMAs.
// Requires M % GROWS == 0 (true for 10000*12 = 120000).
// fuse!=0 applies relu(a + bias[col]) while staging A.
// ---------------------------------------------------------------------------
__global__ __launch_bounds__(256)
void gemm_wmma_bf16(const float* __restrict__ A,
                    const unsigned short* __restrict__ Wt,   // bf16 [n][k]
                    const float* __restrict__ bias,          // null if !fuse
                    float* __restrict__ C, int M, int fuse) {
    __shared__ unsigned short sW[FEAT * LDSW];    // 33792 B
    __shared__ unsigned short sA[GROWS * LDSW];   // 42240 B

    const int tid   = threadIdx.x;
    const int mbase = blockIdx.x * GROWS;

    // stage W into padded LDS
    for (int i = tid; i < FEAT * FEAT; i += 256) {
        int n = i >> 7, k = i & 127;
        sW[n * LDSW + k] = Wt[i];
    }
    // stage A tile: float4 load -> (opt bias+relu) -> 4x bf16 -> 8B LDS store
    const float4* A4 = (const float4*)A;
    const float4* B4 = (const float4*)bias;
    for (int i = tid; i < GROWS * (FEAT / 4); i += 256) {
        int r = i >> 5, q = i & 31;              // q: float4 index in row
        float4 v = A4[(size_t)(mbase + r) * (FEAT / 4) + q];
        if (fuse) {
            float4 bb = B4[q];
            v.x = fmaxf(v.x + bb.x, 0.f);
            v.y = fmaxf(v.y + bb.y, 0.f);
            v.z = fmaxf(v.z + bb.z, 0.f);
            v.w = fmaxf(v.w + bb.w, 0.f);
        }
        unsigned short h[4] = {f2bf(v.x), f2bf(v.y), f2bf(v.z), f2bf(v.w)};
        *(unsigned long long*)&sA[r * LDSW + q * 4] =
            *(const unsigned long long*)h;
    }
    __syncthreads();

    const int wave = tid >> 5;        // N tile 0..7
    const int lane = tid & 31;
    const int half = lane >> 4;
    const int lm   = lane & 15;

    // B fragments: register-resident for all 10 M tiles
    FragAB b[4];
    #pragma unroll
    for (int kb = 0; kb < 4; ++kb) {
        #pragma unroll
        for (int g = 0; g < 8; ++g) {
            int kc = kb * 32 + half * 16 + 2 * g;
            b[kb].u[g] = *(const unsigned*)&sW[(wave * 16 + lm) * LDSW + kc];
        }
    }

    const int col = wave * 16 + lm;
    #pragma unroll
    for (int t = 0; t < GT; t += 2) {
        FragC acc0; acc0.v = (v8f){0.f,0.f,0.f,0.f,0.f,0.f,0.f,0.f};
        FragC acc1; acc1.v = (v8f){0.f,0.f,0.f,0.f,0.f,0.f,0.f,0.f};
        #pragma unroll
        for (int kb = 0; kb < 4; ++kb) {
            FragAB a0, a1;
            #pragma unroll
            for (int g = 0; g < 8; ++g) {
                int blk = g >> 2, w = g & 3;
                int ka = kb * 32 + half * 16 + 2 * g;
                a0.u[g] = *(const unsigned*)&sA[((t + 0) * 16 + lm) * LDSW + ka];
                a1.u[g] = *(const unsigned*)&sA[((t + 1) * 16 + lm) * LDSW + ka];
            }
            acc0.v = __builtin_amdgcn_wmma_f32_16x16x32_bf16(
                false, a0.v, false, b[kb].v, (short)0, acc0.v, false, false);
            acc1.v = __builtin_amdgcn_wmma_f32_16x16x32_bf16(
                false, a1.v, false, b[kb].v, (short)0, acc1.v, false, false);
        }
        // unguarded stores: base + r*FEAT become immediate-offset clauses
        float* c0 = C + (size_t)(mbase + (t + 0) * 16 + half * 8) * FEAT + col;
        float* c1 = C + (size_t)(mbase + (t + 1) * 16 + half * 8) * FEAT + col;
        #pragma unroll
        for (int r = 0; r < 8; ++r) c0[r * FEAT] = acc0.f[r];
        #pragma unroll
        for (int r = 0; r < 8; ++r) c1[r * FEAT] = acc1.f[r];
    }
}

// ---------------------------------------------------------------------------
// Scatter: one block per (edge | self-loop). 128-bit gathers of h[src]
// (L2 resident), norm-scaled f32 atomic adds into agg[dst].
// ---------------------------------------------------------------------------
__global__ __launch_bounds__(128)
void scatter_kernel(const int* __restrict__ ei, int E, int nNodes,
                    const float* __restrict__ dis,
                    const float* __restrict__ h,
                    float* __restrict__ agg) {
    int job = blockIdx.x;
    int s, d;
    if (job < E) { s = ei[job]; d = ei[E + job]; }
    else         { s = d = job - E; }
    float nrm = dis[s] * dis[d];

    const float4* hp = (const float4*)(h + (size_t)s * ROWLEN);
    float* ap = agg + (size_t)d * ROWLEN;
    int t = threadIdx.x;
    #pragma unroll
    for (int j = 0; j < 3; ++j) {
        int idx = t + j * 128;          // 384 float4 per row
        float4 v = hp[idx];
        float* o = ap + idx * 4;
        atomicAdd(o + 0, v.x * nrm);
        atomicAdd(o + 1, v.y * nrm);
        atomicAdd(o + 2, v.z * nrm);
        atomicAdd(o + 3, v.w * nrm);
    }
}

// ---------------------------------------------------------------------------
extern "C" void kernel_launch(void* const* d_in, const int* in_sizes, int n_in,
                              void* d_out, int out_size, void* d_ws, size_t ws_size,
                              hipStream_t stream) {
    const float* x  = (const float*)d_in[0];
    const int*   ei = (const int*)  d_in[1];
    const float* W1 = (const float*)d_in[2];
    const float* b1 = (const float*)d_in[3];
    const float* W2 = (const float*)d_in[4];
    const float* b2 = (const float*)d_in[5];
    float* out = (float*)d_out;

    const int nNodes = in_sizes[0] / ROWLEN;       // 10000
    const int E      = in_sizes[1] / 2;            // 160000
    const int M      = nNodes * TSTEPS;            // 120000 rows (multiple of 160)
    const int NF     = M * FEAT;                   // 15,360,000 floats
    const int jobs   = E + nNodes;

    // workspace layout (256B aligned): dis | Wt | hbuf | agg  (~123 MB)
    char* ws = (char*)d_ws;
    size_t off = 0;
    float* dis = (float*)(ws + off);                  off += ((size_t)nNodes * 4 + 255) & ~(size_t)255;
    unsigned short* Wt = (unsigned short*)(ws + off); off += (FEAT * FEAT * 2 + 255) & ~255;
    float* hbuf = (float*)(ws + off);                 off += ((size_t)NF * 4 + 255) & ~(size_t)255;
    float* agg  = (float*)(ws + off);                 // NF floats

    const int B256 = 256;
    dim3 gN((nNodes + B256 - 1) / B256);
    dim3 gJ((jobs + B256 - 1) / B256);
    dim3 gNF((NF + B256 - 1) / B256);
    dim3 gW((FEAT * FEAT + B256 - 1) / B256);
    dim3 gGemm(M / GROWS);                          // 750 blocks (exact)
    dim3 gScat(jobs);

    // --- symmetric normalization ---
    zero_f32<<<gN, B256, 0, stream>>>(dis, nNodes);
    degree_kernel<<<gJ, B256, 0, stream>>>(ei, E, nNodes, dis);
    rsqrt_kernel<<<gN, B256, 0, stream>>>(dis, nNodes);

    // --- layer 1: h = x @ W1 ; agg1 = scatter(norm * h) ---
    convert_w<<<gW, B256, 0, stream>>>(W1, Wt);
    gemm_wmma_bf16<<<gGemm, B256, 0, stream>>>(x, Wt, (const float*)nullptr,
                                               hbuf, M, 0);
    zero_f32<<<gNF, B256, 0, stream>>>(agg, NF);
    scatter_kernel<<<gScat, 128, 0, stream>>>(ei, E, nNodes, dis, hbuf, agg);

    // --- layer 2: h2 = relu(agg1 + b1) @ W2 (activation fused into staging) ---
    convert_w<<<gW, B256, 0, stream>>>(W2, Wt);
    gemm_wmma_bf16<<<gGemm, B256, 0, stream>>>(agg, Wt, b1, hbuf, M, 1);
    zero_f32<<<gNF, B256, 0, stream>>>(out, NF);
    scatter_kernel<<<gScat, 128, 0, stream>>>(ei, E, nNodes, dis, hbuf, out);
    bias_relu<<<gNF, B256, 0, stream>>>(out, b2, NF);
}
